// CrossAttention_46480136077566
// MI455X (gfx1250) — matrix operand order repaired and checked by reading, not compiled, not run on hardware
//
#include <hip/hip_runtime.h>

typedef __attribute__((ext_vector_type(16))) __bf16 bf16x16;
typedef __attribute__((ext_vector_type(8)))  float  f32x8;
typedef __attribute__((ext_vector_type(4)))  unsigned short u16x4;
typedef __attribute__((ext_vector_type(16))) unsigned short u16x16;

__device__ __forceinline__ unsigned short f2bf(float f) {
  union { float f; unsigned u; } c; c.f = f;
  unsigned u = c.u;
  u += 0x7FFFu + ((u >> 16) & 1u);           // round-to-nearest-even
  return (unsigned short)(u >> 16);
}

__device__ __forceinline__ f32x8 wmma_bf16x32(u16x16 a, u16x16 b, f32x8 c) {
  return __builtin_amdgcn_wmma_f32_16x16x32_bf16(
      false, __builtin_bit_cast(bf16x16, a),
      false, __builtin_bit_cast(bf16x16, b),
      (short)0, c, false, false);
}

// A fragment (16x32 bf16) from row-major storage, leading dim ld (elements).
// ISA layout: lanes 0-15 hold K 0..7 / 16..23, lanes 16-31 hold K 8..15 / 24..31.
__device__ __forceinline__ u16x16 load_a_frag(const unsigned short* p0, int ld, int lane) {
  int r = lane & 15;
  int kh = (lane >> 4) << 3;
  const unsigned short* p = p0 + r * ld + kh;
  u16x4 a0 = *(const u16x4*)(p);
  u16x4 a1 = *(const u16x4*)(p + 4);
  u16x4 a2 = *(const u16x4*)(p + 16);
  u16x4 a3 = *(const u16x4*)(p + 20);
  u16x16 o;
#pragma unroll
  for (int i = 0; i < 4; ++i) { o[i] = a0[i]; o[4+i] = a1[i]; o[8+i] = a2[i]; o[12+i] = a3[i]; }
  return o;
}

// B fragment (32x16 bf16) from N-major storage src[n*ld + k].
// ISA layout: lanes 0-15 hold K 0..7 / 16..23, lanes 16-31 hold K 8..15 / 24..31.
__device__ __forceinline__ u16x16 load_b_frag(const unsigned short* p0, int ld, int lane) {
  int n = lane & 15;
  int kh = (lane >> 4) << 3;
  const unsigned short* p = p0 + n * ld + kh;
  u16x4 b0 = *(const u16x4*)(p);
  u16x4 b1 = *(const u16x4*)(p + 4);
  u16x4 b2 = *(const u16x4*)(p + 16);
  u16x4 b3 = *(const u16x4*)(p + 20);
  u16x16 o;
#pragma unroll
  for (int i = 0; i < 4; ++i) { o[i] = b0[i]; o[4+i] = b1[i]; o[8+i] = b2[i]; o[12+i] = b3[i]; }
  return o;
}

// C[M,N] = alpha * A[M,K] @ B[K,N]; A,B bf16 raw. MODE 0: store f32; 1: f32 +=; 2: store bf16.
// Ping-pong double-buffered LDS: one barrier per K-step, next tile prefetched into
// registers while WMMAs run on the current tile.
template<int MODE>
__global__ __launch_bounds__(256) void gemm_wmma(
    const unsigned short* __restrict__ A, const unsigned short* __restrict__ B,
    void* __restrict__ Cp, int M, int N, int K, int lda, int ldb, int ldc, float alpha) {
  constexpr int BM = 128, BN = 64, BK = 32, LS = BK + 8;
  __shared__ unsigned short As[2][BM * LS];   // [m][k]
  __shared__ unsigned short Bs[2][BN * LS];   // [n][k] (transposed)
  const int tid = threadIdx.x;
  const int lane = tid & 31, wave = tid >> 5;
  const int wm = wave >> 1, wn = wave & 1;
  const int m0 = blockIdx.x * BM, n0 = blockIdx.y * BN;

  u16x4 ra[4], rb[2];
  auto g2r = [&](int k0) {
#pragma unroll
    for (int i = 0; i < 4; ++i) {                 // A: 128x32 = 1024 u16x4 chunks
      int c = tid + i * 256;
      int row = c >> 3, k4 = (c & 7) << 2;
      ra[i] = *(const u16x4*)&A[(size_t)(m0 + row) * lda + k0 + k4];
    }
#pragma unroll
    for (int i = 0; i < 2; ++i) {                 // B: 32x64
      int c = tid + i * 256;
      int k = c >> 4, n4 = (c & 15) << 2;
      rb[i] = *(const u16x4*)&B[(size_t)(k0 + k) * ldb + n0 + n4];
    }
  };
  auto r2s = [&](int buf) {
#pragma unroll
    for (int i = 0; i < 4; ++i) {
      int c = tid + i * 256;
      int row = c >> 3, k4 = (c & 7) << 2;
      *(u16x4*)&As[buf][row * LS + k4] = ra[i];
    }
#pragma unroll
    for (int i = 0; i < 2; ++i) {
      int c = tid + i * 256;
      int k = c >> 4, n4 = (c & 15) << 2;
#pragma unroll
      for (int t = 0; t < 4; ++t) Bs[buf][(n4 + t) * LS + k] = rb[i][t];
    }
  };

  f32x8 acc[2][2];
#pragma unroll
  for (int i = 0; i < 2; ++i)
#pragma unroll
    for (int j = 0; j < 2; ++j)
#pragma unroll
      for (int e = 0; e < 8; ++e) acc[i][j][e] = 0.0f;

  const int nk = K / BK;
  g2r(0);
  r2s(0);
  __syncthreads();
  for (int kt = 0; kt < nk; ++kt) {
    const int cur = kt & 1;
    if (kt + 1 < nk) g2r((kt + 1) * BK);
    u16x16 af0 = load_a_frag(&As[cur][(wm * 32) * LS], LS, lane);
    u16x16 af1 = load_a_frag(&As[cur][(wm * 32 + 16) * LS], LS, lane);
    u16x16 bf0 = load_b_frag(&Bs[cur][(wn * 32) * LS], LS, lane);
    u16x16 bf1 = load_b_frag(&Bs[cur][(wn * 32 + 16) * LS], LS, lane);
    acc[0][0] = wmma_bf16x32(af0, bf0, acc[0][0]);
    acc[0][1] = wmma_bf16x32(af0, bf1, acc[0][1]);
    acc[1][0] = wmma_bf16x32(af1, bf0, acc[1][0]);
    acc[1][1] = wmma_bf16x32(af1, bf1, acc[1][1]);
    if (kt + 1 < nk) r2s(cur ^ 1);
    __syncthreads();
  }

  const int half = lane >> 4, cn = lane & 15;
#pragma unroll
  for (int ms = 0; ms < 2; ++ms)
#pragma unroll
    for (int ns = 0; ns < 2; ++ns) {
      int gr0 = m0 + wm * 32 + ms * 16 + half * 8;
      int gc = n0 + wn * 32 + ns * 16 + cn;
#pragma unroll
      for (int g = 0; g < 8; ++g) {
        float v = acc[ms][ns][g] * alpha;
        size_t idx = (size_t)(gr0 + g) * ldc + gc;
        if (MODE == 0)       ((float*)Cp)[idx] = v;
        else if (MODE == 1)  ((float*)Cp)[idx] += v;
        else                 ((unsigned short*)Cp)[idx] = f2bf(v);
      }
    }
}

// g = silu(h[:, n0+4096..]) * h[:, n0..] where h = A @ Wff1. Both column halves
// computed in one block so the SwiGLU epilogue fuses without storing h.
__global__ __launch_bounds__(256) void gemm_ff1_swiglu(
    const unsigned short* __restrict__ A, const unsigned short* __restrict__ B,
    unsigned short* __restrict__ G, int K, int lda, int ldb, int ldg) {
  constexpr int BM = 128, BN = 64, BK = 32, LS = BK + 8, GOFF = 4096;
  __shared__ unsigned short As[2][BM * LS];
  __shared__ unsigned short Bsa[2][BN * LS];
  __shared__ unsigned short Bsg[2][BN * LS];
  const int tid = threadIdx.x, lane = tid & 31, wave = tid >> 5;
  const int wm = wave >> 1, wn = wave & 1;
  const int m0 = blockIdx.x * BM, n0 = blockIdx.y * BN;

  u16x4 ra[4], rba[2], rbg[2];
  auto g2r = [&](int k0) {
#pragma unroll
    for (int i = 0; i < 4; ++i) {
      int c = tid + i * 256;
      int row = c >> 3, k4 = (c & 7) << 2;
      ra[i] = *(const u16x4*)&A[(size_t)(m0 + row) * lda + k0 + k4];
    }
#pragma unroll
    for (int i = 0; i < 2; ++i) {
      int c = tid + i * 256;
      int k = c >> 4, n4 = (c & 15) << 2;
      rba[i] = *(const u16x4*)&B[(size_t)(k0 + k) * ldb + n0 + n4];
      rbg[i] = *(const u16x4*)&B[(size_t)(k0 + k) * ldb + GOFF + n0 + n4];
    }
  };
  auto r2s = [&](int buf) {
#pragma unroll
    for (int i = 0; i < 4; ++i) {
      int c = tid + i * 256;
      int row = c >> 3, k4 = (c & 7) << 2;
      *(u16x4*)&As[buf][row * LS + k4] = ra[i];
    }
#pragma unroll
    for (int i = 0; i < 2; ++i) {
      int c = tid + i * 256;
      int k = c >> 4, n4 = (c & 15) << 2;
#pragma unroll
      for (int t = 0; t < 4; ++t) {
        Bsa[buf][(n4 + t) * LS + k] = rba[i][t];
        Bsg[buf][(n4 + t) * LS + k] = rbg[i][t];
      }
    }
  };

  f32x8 aa[2][2], ag[2][2];
#pragma unroll
  for (int i = 0; i < 2; ++i)
#pragma unroll
    for (int j = 0; j < 2; ++j)
#pragma unroll
      for (int e = 0; e < 8; ++e) { aa[i][j][e] = 0.0f; ag[i][j][e] = 0.0f; }

  const int nk = K / BK;
  g2r(0);
  r2s(0);
  __syncthreads();
  for (int kt = 0; kt < nk; ++kt) {
    const int cur = kt & 1;
    if (kt + 1 < nk) g2r((kt + 1) * BK);
    u16x16 af0 = load_a_frag(&As[cur][(wm * 32) * LS], LS, lane);
    u16x16 af1 = load_a_frag(&As[cur][(wm * 32 + 16) * LS], LS, lane);
    u16x16 ba0 = load_b_frag(&Bsa[cur][(wn * 32) * LS], LS, lane);
    u16x16 ba1 = load_b_frag(&Bsa[cur][(wn * 32 + 16) * LS], LS, lane);
    u16x16 bg0 = load_b_frag(&Bsg[cur][(wn * 32) * LS], LS, lane);
    u16x16 bg1 = load_b_frag(&Bsg[cur][(wn * 32 + 16) * LS], LS, lane);
    aa[0][0] = wmma_bf16x32(af0, ba0, aa[0][0]);
    aa[0][1] = wmma_bf16x32(af0, ba1, aa[0][1]);
    aa[1][0] = wmma_bf16x32(af1, ba0, aa[1][0]);
    aa[1][1] = wmma_bf16x32(af1, ba1, aa[1][1]);
    ag[0][0] = wmma_bf16x32(af0, bg0, ag[0][0]);
    ag[0][1] = wmma_bf16x32(af0, bg1, ag[0][1]);
    ag[1][0] = wmma_bf16x32(af1, bg0, ag[1][0]);
    ag[1][1] = wmma_bf16x32(af1, bg1, ag[1][1]);
    if (kt + 1 < nk) r2s(cur ^ 1);
    __syncthreads();
  }

  const int half = lane >> 4, cn = lane & 15;
#pragma unroll
  for (int ms = 0; ms < 2; ++ms)
#pragma unroll
    for (int ns = 0; ns < 2; ++ns) {
      int gr0 = m0 + wm * 32 + ms * 16 + half * 8;
      int gc = n0 + wn * 32 + ns * 16 + cn;
#pragma unroll
      for (int g = 0; g < 8; ++g) {
        float a = aa[ms][ns][g];
        float xg = ag[ms][ns][g];
        float sg = xg * __builtin_amdgcn_rcpf(1.0f + __expf(-xg));
        G[(size_t)(gr0 + g) * ldg + gc] = f2bf(a * sg);
      }
    }
}

// Multi-query flash attention: Q (b*2048,1024) bf16 prescaled, KV (b*2048,128)
// bf16 as [k|v]. Each wave: 16 q rows, online softmax, j streamed in 32-tiles,
// K/V tiles double-buffered in LDS (one barrier per tile).
__global__ __launch_bounds__(256) void flash_attn_mqa(
    const unsigned short* __restrict__ Q,
    const unsigned short* __restrict__ KV,
    const int* __restrict__ mask,
    unsigned short* __restrict__ O) {
  constexpr int LK = 72, LV = 40, LP = 40;
  __shared__ unsigned short Kt[2][32 * LK];   // [j][d]
  __shared__ unsigned short Vt[2][64 * LV];   // [d][j] (transposed)
  __shared__ unsigned short Ps[8][16 * LP];   // per-wave probs [r][j]
  const int tid = threadIdx.x, lane = tid & 31, wave = tid >> 5;
  const int b = blockIdx.z, h = blockIdx.y;
  const int q0 = blockIdx.x * 128 + wave * 16;
  const size_t qrow = (size_t)(b * 2048 + q0);

  u16x16 qf0 = load_a_frag(Q + qrow * 1024 + h * 64, 1024, lane);
  u16x16 qf1 = load_a_frag(Q + qrow * 1024 + h * 64 + 32, 1024, lane);

  u16x4 rk[2], rv[2];
  auto ldkv = [&](int j0) {
#pragma unroll
    for (int i = 0; i < 2; ++i) {
      int c = tid + i * 256;
      int j = c >> 4, d4 = (c & 15) << 2;
      size_t base = (size_t)(b * 2048 + j0 + j) * 128;
      rk[i] = *(const u16x4*)&KV[base + d4];
      rv[i] = *(const u16x4*)&KV[base + 64 + d4];
    }
  };
  auto stkv = [&](int buf) {
#pragma unroll
    for (int i = 0; i < 2; ++i) {
      int c = tid + i * 256;
      int j = c >> 4, d4 = (c & 15) << 2;
      *(u16x4*)&Kt[buf][j * LK + d4] = rk[i];
#pragma unroll
      for (int t = 0; t < 4; ++t) Vt[buf][(d4 + t) * LV + j] = rv[i][t];
    }
  };

  f32x8 oacc[4];
  float mrow[8], lrow[8];
#pragma unroll
  for (int t = 0; t < 4; ++t)
#pragma unroll
    for (int e = 0; e < 8; ++e) oacc[t][e] = 0.0f;
#pragma unroll
  for (int g = 0; g < 8; ++g) { mrow[g] = -3.0e38f; lrow[g] = 0.0f; }

  ldkv(0);
  stkv(0);
  __syncthreads();
  for (int jt = 0; jt < 64; ++jt) {
    const int j0 = jt * 32;
    const int cur = jt & 1;
    if (jt + 1 < 64) ldkv(j0 + 32);

    f32x8 s0, s1;
#pragma unroll
    for (int e = 0; e < 8; ++e) { s0[e] = 0.0f; s1[e] = 0.0f; }
    s0 = wmma_bf16x32(qf0, load_b_frag(&Kt[cur][0], LK, lane), s0);
    s0 = wmma_bf16x32(qf1, load_b_frag(&Kt[cur][32], LK, lane), s0);
    s1 = wmma_bf16x32(qf0, load_b_frag(&Kt[cur][16 * LK], LK, lane), s1);
    s1 = wmma_bf16x32(qf1, load_b_frag(&Kt[cur][16 * LK + 32], LK, lane), s1);

    const int cn = lane & 15;
    const int half = lane >> 4;
    float bias0 = mask[b * 2048 + j0 + cn]      ? 0.0f : -3.0e38f;
    float bias1 = mask[b * 2048 + j0 + 16 + cn] ? 0.0f : -3.0e38f;
#pragma unroll
    for (int e = 0; e < 8; ++e) { s0[e] += bias0; s1[e] += bias1; }

    float sc[8];
#pragma unroll
    for (int g = 0; g < 8; ++g) {
      float v = fmaxf(s0[g], s1[g]);
      v = fmaxf(v, __shfl_xor(v, 1, 16));
      v = fmaxf(v, __shfl_xor(v, 2, 16));
      v = fmaxf(v, __shfl_xor(v, 4, 16));
      v = fmaxf(v, __shfl_xor(v, 8, 16));
      float m2 = fmaxf(mrow[g], v);
      sc[g] = __expf(mrow[g] - m2);
      float p0 = __expf(s0[g] - m2);
      float p1 = __expf(s1[g] - m2);
      s0[g] = p0; s1[g] = p1;
      float rs = p0 + p1;
      rs += __shfl_xor(rs, 1, 16);
      rs += __shfl_xor(rs, 2, 16);
      rs += __shfl_xor(rs, 4, 16);
      rs += __shfl_xor(rs, 8, 16);
      lrow[g] = lrow[g] * sc[g] + rs;
      mrow[g] = m2;
    }
#pragma unroll
    for (int t = 0; t < 4; ++t)
#pragma unroll
      for (int g = 0; g < 8; ++g) oacc[t][g] *= sc[g];

    unsigned short* pw = &Ps[wave][0];          // wave-private: no barrier needed
#pragma unroll
    for (int g = 0; g < 8; ++g) {
      pw[(half * 8 + g) * LP + cn]      = f2bf(s0[g]);
      pw[(half * 8 + g) * LP + 16 + cn] = f2bf(s1[g]);
    }
    u16x16 pf = load_a_frag(pw, LP, lane);
    oacc[0] = wmma_bf16x32(pf, load_b_frag(&Vt[cur][0],       LV, lane), oacc[0]);
    oacc[1] = wmma_bf16x32(pf, load_b_frag(&Vt[cur][16 * LV], LV, lane), oacc[1]);
    oacc[2] = wmma_bf16x32(pf, load_b_frag(&Vt[cur][32 * LV], LV, lane), oacc[2]);
    oacc[3] = wmma_bf16x32(pf, load_b_frag(&Vt[cur][48 * LV], LV, lane), oacc[3]);

    if (jt + 1 < 64) stkv(cur ^ 1);
    __syncthreads();
  }

  const int half = lane >> 4, cn = lane & 15;
#pragma unroll
  for (int g = 0; g < 8; ++g) {
    float rinv = __builtin_amdgcn_rcpf(lrow[g]);
#pragma unroll
    for (int t = 0; t < 4; ++t) {
      float v = oacc[t][g] * rinv;
      O[(qrow + half * 8 + g) * 1024 + h * 64 + t * 16 + cn] = f2bf(v);
    }
  }
}

__global__ __launch_bounds__(256) void layernorm_bf16(
    const float* __restrict__ X, const float* __restrict__ gamma,
    unsigned short* __restrict__ Y, int dim) {
  const int row = blockIdx.x;
  const float* x = X + (size_t)row * dim;
  float s = 0.0f, s2 = 0.0f;
  for (int i = threadIdx.x; i < dim; i += 256) {
    float v = x[i]; s += v; s2 += v * v;
  }
#pragma unroll
  for (int o = 16; o > 0; o >>= 1) {
    s  += __shfl_xor(s, o, 32);
    s2 += __shfl_xor(s2, o, 32);
  }
  __shared__ float red[16];
  const int wave = threadIdx.x >> 5, lane = threadIdx.x & 31;
  if (lane == 0) { red[wave] = s; red[8 + wave] = s2; }
  __syncthreads();
  float ts = 0.0f, ts2 = 0.0f;
#pragma unroll
  for (int i = 0; i < 8; ++i) { ts += red[i]; ts2 += red[8 + i]; }
  float rdim = 1.0f / (float)dim;
  float mu = ts * rdim;
  float var = ts2 * rdim - mu * mu;
  float inv = rsqrtf(var + 1e-5f);
  unsigned short* y = Y + (size_t)row * dim;
  for (int i = threadIdx.x; i < dim; i += 256)
    y[i] = f2bf((x[i] - mu) * inv * gamma[i]);
}

__global__ void cvt_f32_to_bf16(const float* __restrict__ in,
                                unsigned short* __restrict__ out, int n) {
  int i = blockIdx.x * 256 + threadIdx.x;
  if (i < n) out[i] = f2bf(in[i]);
}

extern "C" void kernel_launch(void* const* d_in, const int* in_sizes, int n_in,
                              void* d_out, int out_size, void* d_ws, size_t ws_size,
                              hipStream_t stream) {
  (void)in_sizes; (void)n_in; (void)out_size; (void)ws_size;
  const float* x       = (const float*)d_in[0];
  const float* context = (const float*)d_in[1];
  const int*   mask    = (const int*)d_in[2];
  const float* gamma   = (const float*)d_in[3];
  const float* ctxg    = (const float*)d_in[4];
  const float* Wq      = (const float*)d_in[5];
  const float* Wkv     = (const float*)d_in[6];
  const float* Wout    = (const float*)d_in[7];
  const float* Wff1    = (const float*)d_in[8];
  const float* Wff2    = (const float*)d_in[9];
  float* out = (float*)d_out;

  const int B = 2, N = 2048, D = 1024, H = 16, DH = 64, FF = 4096;
  const int R = B * N;  // 4096 rows

  char* ws = (char*)d_ws;
  auto take = [&](size_t elems) {
    unsigned short* p = (unsigned short*)ws;
    ws += ((elems * 2) + 255) & ~(size_t)255;
    return p;
  };
  unsigned short* xn   = take((size_t)R * D);        // layer_norm(x)        bf16
  unsigned short* cxn  = take((size_t)R * D);        // layer_norm(context)  bf16
  unsigned short* wq   = take((size_t)D * D);
  unsigned short* wkv  = take((size_t)D * 2 * DH);
  unsigned short* wout = take((size_t)D * D);
  unsigned short* wff1 = take((size_t)D * 2 * FF);
  unsigned short* wff2 = take((size_t)FF * D);
  unsigned short* qb   = take((size_t)R * D);        // q, prescaled
  unsigned short* kvb  = take((size_t)R * 2 * DH);   // [k|v]
  unsigned short* ao   = take((size_t)R * D);        // attention out
  unsigned short* gb   = take((size_t)R * FF);       // silu(gate)*a

  auto cvt = [&](const float* src, unsigned short* dst, int n) {
    cvt_f32_to_bf16<<<(n + 255) / 256, 256, 0, stream>>>(src, dst, n);
  };
  cvt(Wq,   wq,   D * D);
  cvt(Wkv,  wkv,  D * 2 * DH);
  cvt(Wout, wout, D * D);
  cvt(Wff1, wff1, D * 2 * FF);
  cvt(Wff2, wff2, FF * D);

  layernorm_bf16<<<R, 256, 0, stream>>>(x, gamma, xn, D);
  layernorm_bf16<<<R, 256, 0, stream>>>(context, ctxg, cxn, D);

  // q = xn @ Wq * dim_head^-0.5 (bf16)
  gemm_wmma<2><<<dim3(R / 128, D / 64), 256, 0, stream>>>(
      xn, wq, qb, R, D, D, D, D, D, 0.125f);
  // kv = cxn @ Wkv (bf16)
  gemm_wmma<2><<<dim3(R / 128, (2 * DH) / 64), 256, 0, stream>>>(
      cxn, wkv, kvb, R, 2 * DH, D, D, 2 * DH, 2 * DH, 1.0f);
  // attention (multi-query flash)
  flash_attn_mqa<<<dim3(N / 128, H, B), 256, 0, stream>>>(qb, kvb, mask, ao);
  // out = attn_out @ Wout (f32 store)
  gemm_wmma<0><<<dim3(R / 128, D / 64), 256, 0, stream>>>(
      ao, wout, out, R, D, D, D, D, D, 1.0f);
  // g = silu(gate) * a, fused dual-tile GEMM (bf16)
  gemm_ff1_swiglu<<<dim3(R / 128, FF / 64), 256, 0, stream>>>(
      xn, wff1, gb, D, D, 2 * FF, FF);
  // out += g @ Wff2 (f32 accumulate)
  gemm_wmma<1><<<dim3(R / 128, D / 64), 256, 0, stream>>>(
      gb, wff2, out, R, D, FF, FF, D, D, 1.0f);
}
